// QuantLinearBase_38019050504445
// MI455X (gfx1250) — 8-bit matrix operand order repaired and checked by reading, compiled, not run on hardware
//
#include <hip/hip_runtime.h>
#include <stdint.h>


#define TOKENS 16384
#define IN_F   4096
#define OUT_F  4096

#define BM 128
#define BN 128
#define BK 64
#define KTILES (IN_F / BK)

typedef int v8i __attribute__((ext_vector_type(8)));

__global__ __launch_bounds__(256)
void quant_act_kernel(const float* __restrict__ x,
                      uint8_t* __restrict__ qx,
                      float* __restrict__ ascale) {
  const int t   = blockIdx.x;
  const int tid = threadIdx.x;
  const float4* row = (const float4*)(x + (size_t)t * IN_F);

  float4 v[4];
  float m = 0.0f;
#pragma unroll
  for (int w = 0; w < 4; ++w) {
    v[w] = row[tid + 256 * w];
    m = fmaxf(m, fmaxf(fmaxf(fabsf(v[w].x), fabsf(v[w].y)),
                       fmaxf(fabsf(v[w].z), fabsf(v[w].w))));
  }
#pragma unroll
  for (int off = 16; off > 0; off >>= 1)
    m = fmaxf(m, __shfl_xor(m, off, 32));

  __shared__ float red[8];
  if ((tid & 31) == 0) red[tid >> 5] = m;
  __syncthreads();
  m = red[0];
#pragma unroll
  for (int w = 1; w < 8; ++w) m = fmaxf(m, red[w]);

  float scale = m * (1.0f / 127.0f);
  if (scale == 0.0f) scale = 1.0f;
  const float inv = 1.0f / scale;
  if (tid == 0) ascale[t] = scale;

  int* qrow = (int*)(qx + (size_t)t * IN_F);
#pragma unroll
  for (int w = 0; w < 4; ++w) {
    int q0 = (int)rintf(v[w].x * inv);
    int q1 = (int)rintf(v[w].y * inv);
    int q2 = (int)rintf(v[w].z * inv);
    int q3 = (int)rintf(v[w].w * inv);
    q0 = min(127, max(-128, q0));
    q1 = min(127, max(-128, q1));
    q2 = min(127, max(-128, q2));
    q3 = min(127, max(-128, q3));
    qrow[tid + 256 * w] = (q0 & 255) | ((q1 & 255) << 8) |
                          ((q2 & 255) << 16) | ((q3 & 255) << 24);
  }
}

__device__ __forceinline__ unsigned int pack2(int q) {
  const int hi = ((q >> 4) & 15) - 8;
  const int lo = (q & 15) - 8;
  return (unsigned int)(hi & 255) | ((unsigned int)(lo & 255) << 8);
}

__global__ __launch_bounds__(256)
void unpack_w_kernel(const int* __restrict__ qw, uint8_t* __restrict__ w8) {
  const size_t gid = (size_t)blockIdx.x * blockDim.x + threadIdx.x;
  const int4 q = ((const int4*)qw)[gid];
  uint2 o;
  o.x = pack2(q.x) | (pack2(q.y) << 16);
  o.y = pack2(q.z) | (pack2(q.w) << 16);
  ((uint2*)w8)[gid] = o;
}

__global__ __launch_bounds__(256)
void gemm_wmma_iu8(const uint8_t* __restrict__ qx,
                   const uint8_t* __restrict__ w8,
                   const float* __restrict__ ascale,
                   const float* __restrict__ wscale,
                   const float* __restrict__ bias,
                   float* __restrict__ out) {
  __shared__ __align__(16) uint8_t sA[2][BM * BK];
  __shared__ __align__(16) uint8_t sB[2][BN * BK];

  const int tid   = threadIdx.x;
  const int lane  = tid & 31;
  const int wid   = tid >> 5;
  const int waveM = (wid & 1) * 64;
  const int waveN = (wid >> 1) * 32;
  const int blockM = blockIdx.y * BM;
  const int blockN = blockIdx.x * BN;

  const int lane16   = lane & 15;
  const int laneHalf = lane >> 4;

  const v8i vzero = {0, 0, 0, 0, 0, 0, 0, 0};
  v8i acc[4][2];
#pragma unroll
  for (int mt = 0; mt < 4; ++mt)
#pragma unroll
    for (int nt = 0; nt < 2; ++nt) acc[mt][nt] = vzero;

  unsigned long long gA[2], gB[2];
  unsigned ldsA[2], ldsB[2];
#pragma unroll
  for (int it = 0; it < 2; ++it) {
    const int c = tid + it * 256;
    const int r = c >> 2;
    const int s = (c & 3) << 4;
    gA[it] = (unsigned long long)(qx + (size_t)(blockM + r) * IN_F + s);
    gB[it] = (unsigned long long)(w8 + (size_t)(blockN + r) * IN_F + s);
    ldsA[it] = (unsigned)(unsigned long long)(void*)&sA[0][r * BK + s];
    ldsB[it] = (unsigned)(unsigned long long)(void*)&sB[0][r * BK + s];
  }

  auto issue_async = [&](int k0, int buf) {
    const unsigned boffA = (unsigned)buf * (BM * BK);
    const unsigned boffB = (unsigned)buf * (BN * BK);
#pragma unroll
    for (int it = 0; it < 2; ++it) {
      const unsigned la = ldsA[it] + boffA;
      const unsigned long long ga = gA[it] + (unsigned long long)k0;
      asm volatile("global_load_async_to_lds_b128 %0, %1, off"
                   :: "v"(la), "v"(ga) : "memory");
      const unsigned lb = ldsB[it] + boffB;
      const unsigned long long gb = gB[it] + (unsigned long long)k0;
      asm volatile("global_load_async_to_lds_b128 %0, %1, off"
                   :: "v"(lb), "v"(gb) : "memory");
    }
  };

  issue_async(0, 0);

  for (int kt = 0; kt < KTILES; ++kt) {
    const int cur = kt & 1;

    if (kt + 1 < KTILES) {
      issue_async((kt + 1) * BK, 1 - cur);
      asm volatile("s_wait_asynccnt 0x4" ::: "memory");
    } else {
      asm volatile("s_wait_asynccnt 0x0" ::: "memory");
    }
    __syncthreads();

    if (kt + 2 < KTILES) {
      const int k2 = (kt + 2) * BK;
      __builtin_prefetch(qx + (size_t)(blockM + (tid >> 1)) * IN_F + k2, 0, 1);
      __builtin_prefetch(w8 + (size_t)(blockN + (tid >> 1)) * IN_F + k2, 0, 1);
    }

    v8i afrag[4];
#pragma unroll
    for (int mt = 0; mt < 4; ++mt) {
      const uint8_t* pa =
          &sA[cur][(waveM + mt * 16 + lane16) * BK + laneHalf * 16];
      v8i a;
#pragma unroll
      for (int c = 0; c < 4; ++c) {
        const unsigned long long d = *(const unsigned long long*)(pa + (c & 1) * 8 + (c >> 1) * 32);
        a[2 * c]     = (int)(unsigned int)(d & 0xffffffffull);
        a[2 * c + 1] = (int)(unsigned int)(d >> 32);
      }
      afrag[mt] = a;
    }

    v8i bfrag[2];
#pragma unroll
    for (int nt = 0; nt < 2; ++nt) {
      const uint8_t* pb =
          &sB[cur][(waveN + nt * 16 + lane16) * BK + laneHalf * 16];
      const int4 lo = *(const int4*)(pb);
      const int4 hi = *(const int4*)(pb + 32);
      v8i b;
      b[0] = lo.x; b[1] = lo.y; b[2] = lo.z; b[3] = lo.w;
      b[4] = hi.x; b[5] = hi.y; b[6] = hi.z; b[7] = hi.w;
      bfrag[nt] = b;
    }

#pragma unroll
    for (int mt = 0; mt < 4; ++mt)
#pragma unroll
      for (int nt = 0; nt < 2; ++nt)
        acc[mt][nt] = __builtin_amdgcn_wmma_i32_16x16x64_iu8(
             true, afrag[mt],  true, bfrag[nt],
            acc[mt][nt],  false,  false);

    __syncthreads();
  }

#pragma unroll
  for (int nt = 0; nt < 2; ++nt) {
    const int n = blockN + waveN + nt * 16 + lane16;
    const float ws = wscale[n];
    const float bv = bias[n];
#pragma unroll
    for (int mt = 0; mt < 4; ++mt) {
      const int mbase = blockM + waveM + mt * 16 + laneHalf * 8;
#pragma unroll
      for (int i = 0; i < 8; ++i) {
        const int m = mbase + i;
        out[(size_t)m * OUT_F + n] =
            (float)acc[mt][nt][i] * (ascale[m] * ws) + bv;
      }
    }
  }
}

extern "C" void kernel_launch(void* const* d_in, const int* in_sizes, int n_in,
                              void* d_out, int out_size, void* d_ws,
                              size_t ws_size, hipStream_t stream) {
  const float* x      = (const float*)d_in[0];
  const int*   qw     = (const int*)d_in[1];
  const float* wscale = (const float*)d_in[2];
  const float* bias   = (const float*)d_in[3];
  float* out = (float*)d_out;

  uint8_t* qx = (uint8_t*)d_ws;
  uint8_t* w8 = qx + (size_t)TOKENS * IN_F;
  float* ascale = (float*)(w8 + (size_t)OUT_F * IN_F);

  quant_act_kernel<<<TOKENS, 256, 0, stream>>>(x, qx, ascale);

  const int n_packed = OUT_F * (IN_F / 2);
  unpack_w_kernel<<<n_packed / 4 / 256, 256, 0, stream>>>(qw, w8);

  dim3 grid(OUT_F / BN, TOKENS / BM);
  gemm_wmma_iu8<<<grid, 256, 0, stream>>>(qx, w8, ascale, wscale, bias, out);
}
